// DecompBlock_18786186953498
// MI455X (gfx1250) — matrix operand order repaired and checked by reading, not compiled, not run on hardware
//
#include <hip/hip_runtime.h>
#include <hip/hip_bf16.h>

typedef __attribute__((ext_vector_type(16))) __bf16 v16bf;
typedef __attribute__((ext_vector_type(8)))  __bf16 v8bf;
typedef __attribute__((ext_vector_type(8)))  float  v8f;

#define L 2048
#define N_DFT 8192
#define SEASONS 4
#define KER 25
#define ROWS 2048              // B*C = 64*32
#define TWO_PI 6.283185307179586f

static __device__ __forceinline__ unsigned short f32_to_bf16(float f) {
    union { float f; unsigned u; } v; v.f = f;
    unsigned r = v.u + 0x7FFFu + ((v.u >> 16) & 1u);   // round-to-nearest-even
    return (unsigned short)(r >> 16);
}

// ---------------------------------------------------------------------------
// Kernel 1: Toeplitz generator  g[s][t] = (1/N) * sum_j d[s][j] cos(2pi (sL+j) t / N)
// LDS cosine table of N_DFT entries; incremental table index (+= t mod N).
// ---------------------------------------------------------------------------
__global__ __launch_bounds__(256) void gen_g_kernel(const float* __restrict__ diag,
                                                    float* __restrict__ g) {
    __shared__ float tab[N_DFT];
    const int tid = threadIdx.x;
    for (int i = tid; i < N_DFT; i += 256)
        tab[i] = cosf((float)i * (TWO_PI / (float)N_DFT));
    __syncthreads();

    const int gid = blockIdx.x * 256 + tid;     // 0 .. SEASONS*L-1
    const int s = gid >> 11;
    const int t = gid & (L - 1);
    const float* ds = diag + s * L;
    unsigned idx = ((unsigned)(s * L) * (unsigned)t) & (N_DFT - 1);
    const unsigned step = (unsigned)t;
    float sum = 0.0f;
    for (int j = 0; j < L; ++j) {
        sum += ds[j] * tab[idx];
        idx = (idx + step) & (N_DFT - 1);
    }
    g[gid] = sum * (1.0f / (float)N_DFT);
}

// ---------------------------------------------------------------------------
// Kernel 2: initialize f32 state and bf16 ping buffer from x
// ---------------------------------------------------------------------------
__global__ __launch_bounds__(256) void init_xr_kernel(const float* __restrict__ x,
                                                      float* __restrict__ xr,
                                                      unsigned short* __restrict__ xbf) {
    const int i4 = blockIdx.x * 256 + threadIdx.x;       // 4M/4 = 1M float4s
    float4 v = ((const float4*)x)[i4];
    ((float4*)xr)[i4] = v;
    union { unsigned short us[4]; uint2 u2; } p;
    p.us[0] = f32_to_bf16(v.x);
    p.us[1] = f32_to_bf16(v.y);
    p.us[2] = f32_to_bf16(v.z);
    p.us[3] = f32_to_bf16(v.w);
    ((uint2*)xbf)[i4] = p.u2;
}

// ---------------------------------------------------------------------------
// Kernel 3: expand Toeplitz generator into full bf16 circulant matrix (symmetric)
// ---------------------------------------------------------------------------
__global__ __launch_bounds__(256) void expand_c_kernel(const float* __restrict__ gs,
                                                       unsigned short* __restrict__ c) {
    const int idx8 = blockIdx.x * 256 + threadIdx.x;     // one 8-wide chunk per thread
    const int k  = idx8 >> 8;                            // row
    const int n0 = (idx8 & 255) << 3;                    // col base
    union { unsigned short us[8]; uint4 u4; } p;
#pragma unroll
    for (int i = 0; i < 8; ++i) {
        int d = k - (n0 + i); d = d < 0 ? -d : d;
        p.us[i] = f32_to_bf16(gs[d]);
    }
    ((uint4*)(c + (size_t)k * L + n0))[0] = p.u4;
}

// ---------------------------------------------------------------------------
// Kernel 4: one season update  xr = xr - tanh(xr_bf @ C_bf)
// 128x128 block tile, 8 waves (4x2), wave tile 32x64, K-step 32, bf16 WMMA.
// Double-buffered LDS fed by CDNA5 async global->LDS DMA (ASYNCcnt).
// Writes f32 state in place (own tile) and next-season bf16 buffer.
// ---------------------------------------------------------------------------
#define LDK 40   // padded LDS row stride (bf16 elements)

// Issue one 128x32 bf16 tile (8 KB) as 512 async 16B transfers: 2 per thread.
static __device__ __forceinline__ void issue_tile_async(
        const unsigned short* __restrict__ gsrc,   // matrix base [2048][2048]
        int tile_row_base, int k0, int tid,
        unsigned short* lds_tile) {
#pragma unroll
    for (int i = 0; i < 2; ++i) {
        const int c   = tid + i * 256;
        const int row = c >> 2;
        const int col = (c & 3) << 3;
        const unsigned short* gp =
            gsrc + (size_t)(tile_row_base + row) * L + k0 + col;
        const unsigned lds_off =
            (unsigned)(uintptr_t)(lds_tile + row * LDK + col);
        asm volatile("global_load_async_to_lds_b128 %0, %1, off"
                     :: "v"(lds_off), "v"(gp) : "memory");
    }
}

__global__ __launch_bounds__(256) void gemm_season_kernel(
        const unsigned short* __restrict__ Abf,   // [2048][2048] bf16, current x_rem
        const unsigned short* __restrict__ Cbf,   // [2048][2048] bf16, circulant (symmetric)
        float* __restrict__ xr,                   // [2048][2048] f32 state (in/out)
        unsigned short* __restrict__ Anext) {     // next-season bf16 buffer
    __shared__ unsigned short As[2][128 * LDK];
    __shared__ unsigned short Bs[2][128 * LDK];

    const int tid  = threadIdx.x;
    const int lane = tid & 31;
    const int wave = tid >> 5;
    const int wm = wave >> 1;          // 0..3
    const int wn = wave & 1;           // 0..1
    const int bm = blockIdx.y * 128;
    const int bn = blockIdx.x * 128;

    v8f acc[2][4];
#pragma unroll
    for (int i = 0; i < 2; ++i)
#pragma unroll
        for (int j = 0; j < 4; ++j)
            acc[i][j] = (v8f)(0.0f);

    const int kb = (lane < 16) ? 0 : 8;    // A fragment K base (lo half)
    const int kc = (lane < 16) ? 0 : 8;    // B fragment K base
    const int ml = wm * 32 + (lane & 15);
    const int nl = wn * 64 + (lane & 15);

    // Prologue: fill buffer 0 asynchronously.
    issue_tile_async(Abf, bm, 0, tid, &As[0][0]);
    issue_tile_async(Cbf, bn, 0, tid, &Bs[0][0]);

    for (int k0 = 0; k0 < L; k0 += 32) {
        const int cur = (k0 >> 5) & 1;

        // Our async fills for buf[cur] done, then make them block-visible.
        asm volatile("s_wait_asynccnt 0" ::: "memory");
        __syncthreads();

        // Prefetch next stage into the other buffer while we compute.
        if (k0 + 32 < L) {
            issue_tile_async(Abf, bm, k0 + 32, tid, &As[cur ^ 1][0]);
            issue_tile_async(Cbf, bn, k0 + 32, tid, &Bs[cur ^ 1][0]);
        }

        const unsigned short* Ab = &As[cur][0];
        const unsigned short* Bb = &Bs[cur][0];

        v16bf a[2], b[4];
#pragma unroll
        for (int wi = 0; wi < 2; ++wi) {
            const v8bf lo = *(const v8bf*)&Ab[(ml + wi * 16) * LDK + kb];
            const v8bf hi = *(const v8bf*)&Ab[(ml + wi * 16) * LDK + kb + 16];
            a[wi] = __builtin_shufflevector(lo, hi, 0,1,2,3,4,5,6,7,
                                                    8,9,10,11,12,13,14,15);
        }
#pragma unroll
        for (int wj = 0; wj < 4; ++wj) {
            const v8bf lo = *(const v8bf*)&Bb[(nl + wj * 16) * LDK + kc];
            const v8bf hi = *(const v8bf*)&Bb[(nl + wj * 16) * LDK + kc + 16];
            b[wj] = __builtin_shufflevector(lo, hi, 0,1,2,3,4,5,6,7,
                                                    8,9,10,11,12,13,14,15);
        }
#pragma unroll
        for (int wi = 0; wi < 2; ++wi)
#pragma unroll
            for (int wj = 0; wj < 4; ++wj)
                acc[wi][wj] = __builtin_amdgcn_wmma_f32_16x16x32_bf16(
                    false, a[wi], false, b[wj], (short)0, acc[wi][wj], false, false);
        // No trailing barrier: a wave's fragment ds_loads complete (s_wait_dscnt
        // before the last WMMA) before it reaches the next iteration's barrier,
        // and the next-stage DMA targets the other buffer.
    }

    // Epilogue: xr_new = xr_old - tanh(Y); write f32 state + bf16 next buffer.
#pragma unroll
    for (int wi = 0; wi < 2; ++wi) {
#pragma unroll
        for (int wj = 0; wj < 4; ++wj) {
            const int ng = bn + wn * 64 + wj * 16 + (lane & 15);
#pragma unroll
            for (int r = 0; r < 8; ++r) {
                const int mg = bm + wm * 32 + wi * 16 + r + ((lane < 16) ? 0 : 8);
                const size_t off = (size_t)mg * L + ng;
                const float y   = acc[wi][wj][r];
                const float v   = xr[off] - tanhf(y);
                xr[off]    = v;
                Anext[off] = f32_to_bf16(v);
            }
        }
    }
}

// ---------------------------------------------------------------------------
// Kernel 5: trend (replicate-pad front, K=25 moving average) + final output
//           out = x - xr + trend
// ---------------------------------------------------------------------------
__global__ __launch_bounds__(256) void trend_final_kernel(const float* __restrict__ x,
                                                          const float* __restrict__ xr,
                                                          float* __restrict__ out) {
    __shared__ float row[L];
    const int m = blockIdx.x;
    const float* xrow = xr + (size_t)m * L;
    for (int i = threadIdx.x; i < L / 4; i += 256)
        ((float4*)row)[i] = ((const float4*)xrow)[i];
    __syncthreads();

    const float r0 = row[0];
    const int t0 = threadIdx.x * 8;
#pragma unroll
    for (int q = 0; q < 8; ++q) {
        const int t = t0 + q;
        float s = 0.0f;
        int lo = t - (KER - 1);
        if (lo < 0) { s = (float)(-lo) * r0; lo = 0; }
        for (int u = lo; u <= t; ++u) s += row[u];
        const float trend = s * (1.0f / (float)KER);
        const size_t off = (size_t)m * L + t;
        out[off] = x[off] - row[t] + trend;
    }
}

// ---------------------------------------------------------------------------
// Host-side orchestration
// ---------------------------------------------------------------------------
extern "C" void kernel_launch(void* const* d_in, const int* in_sizes, int n_in,
                              void* d_out, int out_size, void* d_ws, size_t ws_size,
                              hipStream_t stream) {
    const float* x    = (const float*)d_in[0];   // [64,32,2048] f32
    const float* diag = (const float*)d_in[1];   // [4,2048]     f32
    float* out = (float*)d_out;                  // [64,32,2048] f32

    char* ws = (char*)d_ws;
    float*          g    = (float*)ws;                                   // 32 KB
    float*          xr   = (float*)(ws + 32768);                         // 16 MB f32 state
    unsigned short* bf0  = (unsigned short*)(ws + 32768 + 16777216);     // 8 MB bf16 ping
    unsigned short* bf1  = bf0 + (size_t)ROWS * L;                       // 8 MB bf16 pong
    unsigned short* cmat = bf1 + (size_t)ROWS * L;                       // 8 MB bf16 circulant

    gen_g_kernel<<<dim3((SEASONS * L) / 256), dim3(256), 0, stream>>>(diag, g);
    init_xr_kernel<<<dim3((ROWS * L) / 1024), dim3(256), 0, stream>>>(x, xr, bf0);

    unsigned short* bufs[2] = { bf0, bf1 };
    for (int s = 0; s < SEASONS; ++s) {
        expand_c_kernel<<<dim3((L * L) / (256 * 8)), dim3(256), 0, stream>>>(g + s * L, cmat);
        gemm_season_kernel<<<dim3(L / 128, ROWS / 128), dim3(256), 0, stream>>>(
            bufs[s & 1], cmat, xr, bufs[(s + 1) & 1]);
    }

    trend_final_kernel<<<dim3(ROWS), dim3(256), 0, stream>>>(x, xr, out);
}
